// SharedGNNEncoder_44160853738152
// MI455X (gfx1250) — matrix operand order repaired and checked by reading, not compiled, not run on hardware
//
#include <hip/hip_runtime.h>
#include <hip/hip_bf16.h>
#include <math.h>

#define NN 50000
#define NE 300000
#define EP (NE + NN)          // edges + self loops = 350000
#define NB 500
#define HID 128
#define HEADS 6
#define HC (HEADS * HID)      // 768
#define FEAT 9
#define MT ((NN + 15) / 16)   // 3125 m-tiles
#define NT (HC / 16)          // 48 n-tiles
#define KTMAX 4

typedef __attribute__((ext_vector_type(16))) __bf16 v16bf;
typedef __attribute__((ext_vector_type(8)))  float  v8f;

// ---------- helpers ----------
__device__ __forceinline__ unsigned encf(float f) {
  unsigned u = __float_as_uint(f);
  return (u & 0x80000000u) ? ~u : (u | 0x80000000u);
}
__device__ __forceinline__ float decf(unsigned e) {
  unsigned u = (e & 0x80000000u) ? (e & 0x7FFFFFFFu) : ~e;
  return __uint_as_float(u);
}
__device__ __forceinline__ float waveSum(float v) {
  #pragma unroll
  for (int off = 16; off > 0; off >>= 1) v += __shfl_xor(v, off, 32);
  return v;
}
__device__ __forceinline__ float sigmf(float x) { return 1.f / (1.f + expf(-x)); }

// ---------- edge list with self loops ----------
__global__ void prep_edges_kernel(const int* __restrict__ ei, int* __restrict__ src,
                                  int* __restrict__ dst) {
  int i = blockIdx.x * blockDim.x + threadIdx.x;
  if (i < NE) { src[i] = ei[i]; dst[i] = ei[NE + i]; }
  else if (i < EP) { src[i] = i - NE; dst[i] = i - NE; }
}

// ---------- pack W[K,768] f32 -> bf16 B-fragments ----------
// frag (nt,kt): lane n=lane&15, kb=(lane>>4)*8; j<8: B[kt*32+kb+j] ; j>=8: B[kt*32+16+kb+j-8] (col nt*16+n)
__global__ void pack_w_kernel(const float* __restrict__ W, __bf16* __restrict__ PW,
                              int KT, int Kreal) {
  int tid = blockIdx.x * blockDim.x + threadIdx.x;
  int total = NT * KT * 512;
  if (tid >= total) return;
  int frag = tid / 512, r = tid % 512, lane = r / 16, j = r % 16;
  int nt = frag / KT, kt = frag % KT;
  int n = lane & 15, kb = (lane >> 4) * 8;
  int k = kt * 32 + ((j < 8) ? (kb + j) : (16 + kb + (j - 8)));
  int col = nt * 16 + n;
  float v = (k < Kreal) ? W[(size_t)k * HC + col] : 0.f;
  PW[tid] = (__bf16)v;
}

// ---------- pack X[N,ldx] f32 -> bf16 A-fragments ----------
// frag (mt,kt): lane m=lane&15, kg=(lane>>4)*8; j<8: K=kt*32+kg+j ; j>=8: K=kt*32+16+kg+j-8
__global__ void pack_a_kernel(const float* __restrict__ X, __bf16* __restrict__ PA,
                              int KT, int Kreal, int ldx) {
  int tid = blockIdx.x * blockDim.x + threadIdx.x;
  int total = MT * KT * 512;
  if (tid >= total) return;
  int frag = tid / 512, r = tid % 512, lane = r / 16, j = r % 16;
  int mt = frag / KT, kt = frag % KT;
  int m = lane & 15, kg = (lane >> 4) * 8;
  int k = kt * 32 + ((j < 8) ? (kg + j) : (16 + kg + (j - 8)));
  int row = mt * 16 + m;
  float v = (k < Kreal && row < NN) ? X[(size_t)row * ldx + k] : 0.f;
  PA[tid] = (__bf16)v;
}

// ---------- WMMA GEMM: out[N,768] = A(bf16) @ B(bf16) + bias, f32 accum ----------
// block = 256 (8 waves). Each wave computes a 16x64 strip (1 m-tile x 4 n-tiles):
//   mt = bx*8 + wave, n-group = by. A fragment loaded once per K-step, reused for
//   4 WMMAs. KT is compile-time so all fragment loads use immediate offsets.
template <int KT>
__global__ void gemm_wmma_kernel(const __bf16* __restrict__ PA,
                                 const __bf16* __restrict__ PWl,
                                 const __bf16* __restrict__ PWr,
                                 const float* __restrict__ bl, const float* __restrict__ br,
                                 float* __restrict__ xl, float* __restrict__ xr) {
  int lane = threadIdx.x & 31;
  int wave = threadIdx.x >> 5;
  int mt = blockIdx.x * 8 + wave;
  int ntg = blockIdx.y;             // n-group: 4 tiles = 64 columns
  if (mt >= MT) return;             // uniform per-wave, EXEC stays all-ones for WMMA
  const __bf16* PW = blockIdx.z ? PWr : PWl;
  const float* bias = blockIdx.z ? br : bl;
  float* out = blockIdx.z ? xr : xl;

  const __bf16* pa = PA + (size_t)mt * KT * 512 + lane * 16;
  const __bf16* pw = PW + (size_t)ntg * 4 * KT * 512 + lane * 16;

  v8f acc[4];
  #pragma unroll
  for (int j = 0; j < 4; ++j) acc[j] = (v8f){0.f, 0.f, 0.f, 0.f, 0.f, 0.f, 0.f, 0.f};

  #pragma unroll
  for (int kt = 0; kt < KT; ++kt) {
    v16bf a = *(const v16bf*)(pa + kt * 512);
    #pragma unroll
    for (int j = 0; j < 4; ++j) {
      v16bf b = *(const v16bf*)(pw + (j * KT + kt) * 512);
      acc[j] = __builtin_amdgcn_wmma_f32_16x16x32_bf16(false, a, false, b, (short)0, acc[j],
                                                       false, false);
    }
  }

  int row0 = mt * 16 + ((lane >> 4) * 8);
  int cl = lane & 15;
  #pragma unroll
  for (int j = 0; j < 4; ++j) {
    int col = (ntg * 4 + j) * 16 + cl;
    float bv = bias[col];
    #pragma unroll
    for (int r = 0; r < 8; ++r)
      out[(size_t)(row0 + r) * HC + col] = acc[j][r] + bv;
  }
}

// ---------- generic u32 fill ----------
__global__ void fill_u32_kernel(unsigned* __restrict__ p, unsigned v, int n) {
  int i = blockIdx.x * blockDim.x + threadIdx.x;
  if (i < n) p[i] = v;
}

// ---------- pass A: attention logits + segment max (wave per (edge,head)) ----------
__global__ void edge_logits_kernel(const int* __restrict__ src, const int* __restrict__ dst,
                                   const float* __restrict__ xl, const float* __restrict__ xr,
                                   const float* __restrict__ att, float* __restrict__ logits,
                                   unsigned* __restrict__ emaxE) {
  int lane = threadIdx.x & 31;
  int gw = (blockIdx.x * blockDim.x + threadIdx.x) >> 5;
  if (gw >= EP * HEADS) return;
  int e = gw / HEADS, h = gw % HEADS;
  int s = src[e], d = dst[e];
  const float4 a = *(const float4*)(xl + (size_t)s * HC + h * HID + lane * 4);
  const float4 b = *(const float4*)(xr + (size_t)d * HC + h * HID + lane * 4);
  const float4 w = *(const float4*)(att + h * HID + lane * 4);
  float p = 0.f, m;
  m = a.x + b.x; m = m > 0.f ? m : 0.2f * m; p += m * w.x;
  m = a.y + b.y; m = m > 0.f ? m : 0.2f * m; p += m * w.y;
  m = a.z + b.z; m = m > 0.f ? m : 0.2f * m; p += m * w.z;
  m = a.w + b.w; m = m > 0.f ? m : 0.2f * m; p += m * w.w;
  float tot = waveSum(p);
  if (lane == 0) {
    logits[(size_t)e * HEADS + h] = tot;
    atomicMax(emaxE + (size_t)d * HEADS + h, encf(tot));
  }
}

// ---------- pass B: alpha = exp(e - max), denom accumulate ----------
__global__ void edge_alpha_kernel(const int* __restrict__ dst, const float* __restrict__ logits,
                                  const unsigned* __restrict__ emaxE, float* __restrict__ alpha,
                                  float* __restrict__ denom) {
  int i = blockIdx.x * blockDim.x + threadIdx.x;
  if (i >= EP * HEADS) return;
  int e = i / HEADS, h = i % HEADS;
  int d = dst[e];
  float a = expf(logits[i] - decf(emaxE[(size_t)d * HEADS + h]));
  alpha[i] = a;
  atomicAdd(denom + (size_t)d * HEADS + h, a);
}

// ---------- pass C: out_acc[dst] += mean_h alpha_h/denom_h * xl[src,h,:] (wave per edge) ----------
__global__ void edge_scatter_kernel(const int* __restrict__ src, const int* __restrict__ dst,
                                    const float* __restrict__ xl, const float* __restrict__ alpha,
                                    const float* __restrict__ denom, float* __restrict__ out_acc) {
  int lane = threadIdx.x & 31;
  int e = (blockIdx.x * blockDim.x + threadIdx.x) >> 5;
  if (e >= EP) return;
  int s = src[e], d = dst[e];
  float w[HEADS];
  #pragma unroll
  for (int h = 0; h < HEADS; ++h)
    w[h] = alpha[(size_t)e * HEADS + h] / denom[(size_t)d * HEADS + h] * (1.f / 6.f);
  float a0 = 0.f, a1 = 0.f, a2 = 0.f, a3 = 0.f;
  #pragma unroll
  for (int h = 0; h < HEADS; ++h) {
    float4 v = *(const float4*)(xl + (size_t)s * HC + h * HID + lane * 4);
    a0 += w[h] * v.x; a1 += w[h] * v.y; a2 += w[h] * v.z; a3 += w[h] * v.w;
  }
  float* o = out_acc + (size_t)d * HID + lane * 4;
  atomicAdd(o + 0, a0); atomicAdd(o + 1, a1); atomicAdd(o + 2, a2); atomicAdd(o + 3, a3);
}

// ---------- pass D: +bias, BN scale, ReLU, LayerNorm, optional residual (wave per node) ----------
__global__ void node_post_kernel(const float* __restrict__ out_acc, const float* __restrict__ ab,
                                 const float* __restrict__ bng, const float* __restrict__ bnb,
                                 const float* __restrict__ resid, int add_resid,
                                 float* __restrict__ xout) {
  int lane = threadIdx.x & 31;
  int n = (blockIdx.x * blockDim.x + threadIdx.x) >> 5;
  if (n >= NN) return;
  float v[4];
  #pragma unroll
  for (int i = 0; i < 4; ++i) {
    int c = lane * 4 + i;
    float t = out_acc[(size_t)n * HID + c] + ab[c];
    t = t * bng[c] + bnb[c];
    v[i] = t > 0.f ? t : 0.f;
  }
  float mu = waveSum(v[0] + v[1] + v[2] + v[3]) * (1.f / 128.f);
  float q = 0.f;
  #pragma unroll
  for (int i = 0; i < 4; ++i) { float d = v[i] - mu; q += d * d; }
  q = waveSum(q) * (1.f / 128.f);
  float rs = rsqrtf(q + 1e-5f);
  #pragma unroll
  for (int i = 0; i < 4; ++i) {
    int c = lane * 4 + i;
    float t = (v[i] - mu) * rs;
    if (add_resid) t += resid[(size_t)n * HID + c];
    xout[(size_t)n * HID + c] = t;
  }
}

// ---------- Set2Set LSTM step (block per graph, 512 threads) ----------
__global__ void lstm_kernel(const float* __restrict__ qstar, const float* __restrict__ Wih,
                            const float* __restrict__ Whh, const float* __restrict__ bih,
                            const float* __restrict__ bhh, float* __restrict__ h,
                            float* __restrict__ c) {
  __shared__ float gates[4 * HID];
  __shared__ float hs[HID];
  __shared__ float qs[2 * HID];
  int b = blockIdx.x;
  int t = threadIdx.x;
  if (t < HID) hs[t] = h[b * HID + t];
  if (t < 2 * HID) qs[t] = qstar[b * 2 * HID + t];
  __syncthreads();
  float g = bih[t] + bhh[t];
  for (int k = 0; k < 2 * HID; ++k) g += qs[k] * Wih[(size_t)t * (2 * HID) + k];
  for (int k = 0; k < HID; ++k) g += hs[k] * Whh[(size_t)t * HID + k];
  gates[t] = g;
  __syncthreads();
  if (t < HID) {
    float ig = sigmf(gates[t]);
    float fg = sigmf(gates[HID + t]);
    float gg = tanhf(gates[2 * HID + t]);
    float og = sigmf(gates[3 * HID + t]);
    float cv = fg * c[b * HID + t] + ig * gg;
    c[b * HID + t] = cv;
    h[b * HID + t] = og * tanhf(cv);
  }
}

// ---------- Set2Set attention (wave per node) ----------
__global__ void s2s_logits_kernel(const float* __restrict__ x, const int* __restrict__ batch,
                                  const float* __restrict__ h, float* __restrict__ e_n,
                                  unsigned* __restrict__ emaxB) {
  int lane = threadIdx.x & 31;
  int n = (blockIdx.x * blockDim.x + threadIdx.x) >> 5;
  if (n >= NN) return;
  int b = batch[n];
  float4 xv = *(const float4*)(x + (size_t)n * HID + lane * 4);
  float4 hv = *(const float4*)(h + (size_t)b * HID + lane * 4);
  float tot = waveSum(xv.x * hv.x + xv.y * hv.y + xv.z * hv.z + xv.w * hv.w);
  if (lane == 0) { e_n[n] = tot; atomicMax(emaxB + b, encf(tot)); }
}

__global__ void s2s_alpha_kernel(const int* __restrict__ batch, const float* __restrict__ e_n,
                                 const unsigned* __restrict__ emaxB, float* __restrict__ a_n,
                                 float* __restrict__ denomB) {
  int n = blockIdx.x * blockDim.x + threadIdx.x;
  if (n >= NN) return;
  int b = batch[n];
  float a = expf(e_n[n] - decf(emaxB[b]));
  a_n[n] = a;
  atomicAdd(denomB + b, a);
}

__global__ void s2s_scatter_kernel(const float* __restrict__ x, const int* __restrict__ batch,
                                   const float* __restrict__ a_n, const float* __restrict__ denomB,
                                   float* __restrict__ r) {
  int lane = threadIdx.x & 31;
  int n = (blockIdx.x * blockDim.x + threadIdx.x) >> 5;
  if (n >= NN) return;
  int b = batch[n];
  float w = a_n[n] / denomB[b];
  float4 xv = *(const float4*)(x + (size_t)n * HID + lane * 4);
  float* o = r + (size_t)b * HID + lane * 4;
  atomicAdd(o + 0, w * xv.x); atomicAdd(o + 1, w * xv.y);
  atomicAdd(o + 2, w * xv.z); atomicAdd(o + 3, w * xv.w);
}

__global__ void s2s_qstar_kernel(const float* __restrict__ h, const float* __restrict__ r,
                                 float* __restrict__ qstar) {
  int i = blockIdx.x * blockDim.x + threadIdx.x;
  if (i >= NB * 2 * HID) return;
  int b = i / (2 * HID), k = i % (2 * HID);
  qstar[i] = (k < HID) ? h[b * HID + k] : r[b * HID + (k - HID)];
}

// ---------- final: relu(qs @ Wrep^T + brep) * grep + brep2 ----------
__global__ void final_kernel(const float* __restrict__ qstar, const float* __restrict__ Wrep,
                             const float* __restrict__ brep, const float* __restrict__ grep,
                             const float* __restrict__ brep2, float* __restrict__ out) {
  int i = blockIdx.x * blockDim.x + threadIdx.x;
  if (i >= NB * HID) return;
  int b = i / HID, j = i % HID;
  float s = brep[j];
  for (int k = 0; k < 2 * HID; ++k) s += qstar[b * 2 * HID + k] * Wrep[(size_t)j * (2 * HID) + k];
  s = s > 0.f ? s : 0.f;
  out[i] = s * grep[j] + brep2[j];
}

extern "C" void kernel_launch(void* const* d_in, const int* in_sizes, int n_in,
                              void* d_out, int out_size, void* d_ws, size_t ws_size,
                              hipStream_t stream) {
  (void)in_sizes; (void)n_in; (void)out_size; (void)ws_size;
  const float* x_in = (const float*)d_in[0];
  const int* edge_index = (const int*)d_in[1];
  const int* batch = (const int*)d_in[2];
  const float *Wl[3], *bl[3], *Wr[3], *br[3], *att[3], *ab[3], *bng[3], *bnb[3];
  for (int i = 0; i < 3; ++i) {
    int b0 = 3 + i * 8;
    Wl[i] = (const float*)d_in[b0 + 0]; bl[i] = (const float*)d_in[b0 + 1];
    Wr[i] = (const float*)d_in[b0 + 2]; br[i] = (const float*)d_in[b0 + 3];
    att[i] = (const float*)d_in[b0 + 4]; ab[i] = (const float*)d_in[b0 + 5];
    bng[i] = (const float*)d_in[b0 + 6]; bnb[i] = (const float*)d_in[b0 + 7];
  }
  const float* Wih = (const float*)d_in[27];
  const float* Whh = (const float*)d_in[28];
  const float* bih = (const float*)d_in[29];
  const float* bhh = (const float*)d_in[30];
  const float* Wrep = (const float*)d_in[31];
  const float* brep = (const float*)d_in[32];
  const float* grep = (const float*)d_in[33];
  const float* brep2 = (const float*)d_in[34];

  char* base = (char*)d_ws;
  size_t off = 0;
  auto alloc = [&](size_t bytes) -> void* {
    void* p = base + off;
    off += (bytes + 255) & ~(size_t)255;
    return p;
  };
  int* src = (int*)alloc((size_t)EP * 4);
  int* dst = (int*)alloc((size_t)EP * 4);
  float* xbuf0 = (float*)alloc((size_t)NN * HID * 4);
  float* xbuf1 = (float*)alloc((size_t)NN * HID * 4);
  float* xl = (float*)alloc((size_t)NN * HC * 4);
  float* xr = (float*)alloc((size_t)NN * HC * 4);
  __bf16* packA = (__bf16*)alloc((size_t)MT * KTMAX * 512 * 2);
  __bf16* packWl = (__bf16*)alloc((size_t)NT * KTMAX * 512 * 2);
  __bf16* packWr = (__bf16*)alloc((size_t)NT * KTMAX * 512 * 2);
  float* logits = (float*)alloc((size_t)EP * HEADS * 4);
  float* alpha = (float*)alloc((size_t)EP * HEADS * 4);
  unsigned* emaxE = (unsigned*)alloc((size_t)NN * HEADS * 4);
  float* denom = (float*)alloc((size_t)NN * HEADS * 4);
  float* out_acc = (float*)alloc((size_t)NN * HID * 4);
  float* hS = (float*)alloc((size_t)NB * HID * 4);
  float* cS = (float*)alloc((size_t)NB * HID * 4);
  float* qstar = (float*)alloc((size_t)NB * 2 * HID * 4);
  float* e_n = (float*)alloc((size_t)NN * 4);
  float* a_n = (float*)alloc((size_t)NN * 4);
  unsigned* emaxB = (unsigned*)alloc((size_t)NB * 4);
  float* denomB = (float*)alloc((size_t)NB * 4);
  float* rS = (float*)alloc((size_t)NB * HID * 4);

  prep_edges_kernel<<<(EP + 255) / 256, 256, 0, stream>>>(edge_index, src, dst);

  const float* layer_in[3] = {x_in, xbuf0, xbuf1};
  float* layer_out[3] = {xbuf0, xbuf1, xbuf0};
  const int Kreal[3] = {FEAT, HID, HID};

  for (int li = 0; li < 3; ++li) {
    int KT = (Kreal[li] + 31) / 32;  // 1,4,4
    int wtot = NT * KT * 512;
    pack_w_kernel<<<(wtot + 255) / 256, 256, 0, stream>>>(Wl[li], packWl, KT, Kreal[li]);
    pack_w_kernel<<<(wtot + 255) / 256, 256, 0, stream>>>(Wr[li], packWr, KT, Kreal[li]);
    int atot = MT * KT * 512;
    pack_a_kernel<<<(atot + 255) / 256, 256, 0, stream>>>(layer_in[li], packA, KT, Kreal[li],
                                                          Kreal[li]);
    dim3 gg((MT + 7) / 8, NT / 4, 2);
    if (KT == 1)
      gemm_wmma_kernel<1><<<gg, 256, 0, stream>>>(packA, packWl, packWr, bl[li], br[li], xl, xr);
    else
      gemm_wmma_kernel<4><<<gg, 256, 0, stream>>>(packA, packWl, packWr, bl[li], br[li], xl, xr);

    fill_u32_kernel<<<(NN * HEADS + 255) / 256, 256, 0, stream>>>(emaxE, 0u, NN * HEADS);
    fill_u32_kernel<<<(NN * HEADS + 255) / 256, 256, 0, stream>>>((unsigned*)denom, 0u, NN * HEADS);
    fill_u32_kernel<<<(NN * HID + 255) / 256, 256, 0, stream>>>((unsigned*)out_acc, 0u, NN * HID);

    int ehWaves = EP * HEADS;
    edge_logits_kernel<<<(ehWaves + 7) / 8, 256, 0, stream>>>(src, dst, xl, xr, att[li], logits,
                                                              emaxE);
    edge_alpha_kernel<<<(ehWaves + 255) / 256, 256, 0, stream>>>(dst, logits, emaxE, alpha, denom);
    edge_scatter_kernel<<<(EP + 7) / 8, 256, 0, stream>>>(src, dst, xl, alpha, denom, out_acc);
    node_post_kernel<<<(NN + 7) / 8, 256, 0, stream>>>(out_acc, ab[li], bng[li], bnb[li],
                                                       layer_in[li], (li == 1) ? 1 : 0,
                                                       layer_out[li]);
  }

  const float* xf = layer_out[2];
  fill_u32_kernel<<<(NB * HID + 255) / 256, 256, 0, stream>>>((unsigned*)hS, 0u, NB * HID);
  fill_u32_kernel<<<(NB * HID + 255) / 256, 256, 0, stream>>>((unsigned*)cS, 0u, NB * HID);
  fill_u32_kernel<<<(NB * 2 * HID + 255) / 256, 256, 0, stream>>>((unsigned*)qstar, 0u,
                                                                  NB * 2 * HID);
  for (int step = 0; step < 3; ++step) {
    lstm_kernel<<<NB, 4 * HID, 0, stream>>>(qstar, Wih, Whh, bih, bhh, hS, cS);
    fill_u32_kernel<<<(NB + 255) / 256, 256, 0, stream>>>(emaxB, 0u, NB);
    fill_u32_kernel<<<(NB + 255) / 256, 256, 0, stream>>>((unsigned*)denomB, 0u, NB);
    fill_u32_kernel<<<(NB * HID + 255) / 256, 256, 0, stream>>>((unsigned*)rS, 0u, NB * HID);
    s2s_logits_kernel<<<(NN + 7) / 8, 256, 0, stream>>>(xf, batch, hS, e_n, emaxB);
    s2s_alpha_kernel<<<(NN + 255) / 256, 256, 0, stream>>>(batch, e_n, emaxB, a_n, denomB);
    s2s_scatter_kernel<<<(NN + 7) / 8, 256, 0, stream>>>(xf, batch, a_n, denomB, rS);
    s2s_qstar_kernel<<<(NB * 2 * HID + 255) / 256, 256, 0, stream>>>(hS, rS, qstar);
  }
  final_kernel<<<(NB * HID + 255) / 256, 256, 0, stream>>>(qstar, Wrep, brep, grep, brep2,
                                                           (float*)d_out);
}
